// LSTMModel_39934605918644
// MI455X (gfx1250) — matrix operand order repaired and checked by reading, not compiled, not run on hardware
//
#include <hip/hip_runtime.h>
#include <hip/hip_bf16.h>

typedef __attribute__((ext_vector_type(16))) _Float16 v16h;
typedef __attribute__((ext_vector_type(8)))  _Float16 v8h;
typedef __attribute__((ext_vector_type(4)))  _Float16 v4h;
typedef __attribute__((ext_vector_type(8)))  float    v8f;
typedef __attribute__((ext_vector_type(4)))  float    v4f;

#define B_ 1024
#define T_ 256
#define I_ 64
#define H_ 128
#define G_ 512   // 4*H

// gfx1250 has native V_TANH_F32 (TRANS32 op); fall back to ocml if the
// builtin is not exposed by this toolchain.
__device__ __forceinline__ float fast_tanh(float x) {
#if __has_builtin(__builtin_amdgcn_tanhf)
  return __builtin_amdgcn_tanhf(x);
#else
  return tanhf(x);
#endif
}
__device__ __forceinline__ float fast_sigmoid(float x) {
  return 0.5f + 0.5f * fast_tanh(0.5f * x);
}

__device__ __forceinline__ v16h cat8(v8h lo, v8h hi) {
  return __builtin_shufflevector(lo, hi, 0,1,2,3,4,5,6,7,8,9,10,11,12,13,14,15);
}

// A operand (16xK row-major tile in LDS) -> v_wmma 16x16x32 f16 A layout:
// lane&15 = row M; lane>>4 selects K-halves {0..7,16..23} vs {8..15,24..31}.
__device__ __forceinline__ v16h load_a(const _Float16* base, int stride, int k0, int lane) {
  int m  = lane & 15;
  int hs = lane >> 4;
  const _Float16* p = base + m * stride + k0 + hs * 8;
  v8h lo = *(const v8h*)(p);        // K = k0 + hs*8 + 0..7
  v8h hi = *(const v8h*)(p + 16);   // K = k0 + 16 + hs*8 + 0..7
  return cat8(lo, hi);
}

// B operand: W stored gate-major [G][K] (i.e. W^T for the GEMM) in LDS.
// lane&15 = column N; lane>>4 selects K-halves {0..7,16..23} vs {8..15,24..31}.
__device__ __forceinline__ v16h load_b(const _Float16* W, int K, int n0, int k0, int lane) {
  int n  = n0 + (lane & 15);
  int hs = lane >> 4;
  const _Float16* p = W + n * K + k0 + hs * 8;
  v8h lo = *(const v8h*)(p);
  v8h hi = *(const v8h*)(p + 16);
  return cat8(lo, hi);
}

// One LSTM layer. Each block owns 16 batch rows for the full time loop.
// IN_K: input width (64 for layer0, 128 for layer1). IN_F16: input tensor dtype.
// WRITE_SEQ: write full h sequence (f16) vs only final h (f32).
template <int IN_K, bool IN_F16, bool WRITE_SEQ>
__global__ __launch_bounds__(256)
void lstm_layer_kernel(const float* __restrict__ inF32,
                       const _Float16* __restrict__ inF16,
                       const _Float16* __restrict__ WihG,   // [G_][IN_K] f16
                       const _Float16* __restrict__ WhhG,   // [G_][H_]  f16
                       const float* __restrict__ biasG,     // [G_]
                       _Float16* __restrict__ seqOut,       // [B][T][H] f16
                       float* __restrict__ lastOut) {       // [B][H] f32
  __shared__ _Float16 WihS[G_ * IN_K];       // 64KB or 128KB
  __shared__ _Float16 WhhS[G_ * H_];         // 128KB
  __shared__ _Float16 xs[2][16 * IN_K];      // double-buffered input tile
  __shared__ _Float16 h_sh[16 * H_];         // recurrent state (A operand)
  __shared__ float    gact[4 * 16 * H_];     // activated gates i,f,g,o (32KB)

  const int tid  = threadIdx.x;
  const int lane = tid & 31;
  const int wave = tid >> 5;          // 8 waves
  const int b0   = blockIdx.x * 16;   // batch tile base

  // ---- park weights in LDS for the whole time loop ----
  for (int i = tid; i < G_ * IN_K / 8; i += 256)
    ((v8h*)WihS)[i] = ((const v8h*)WihG)[i];
  for (int i = tid; i < G_ * H_ / 8; i += 256)
    ((v8h*)WhhS)[i] = ((const v8h*)WhhG)[i];

  v8h zero8 = {};
  for (int i = tid; i < 16 * H_ / 8; i += 256) ((v8h*)h_sh)[i] = zero8;

  // per-tile bias, constant over time: column N = nt*16 + lane%16
  float breg[4];
#pragma unroll
  for (int q = 0; q < 4; ++q)
    breg[q] = biasG[(wave * 4 + q) * 16 + (lane & 15)];

  float creg[8];
#pragma unroll
  for (int r = 0; r < 8; ++r) creg[r] = 0.0f;
  const int cm = tid >> 4;           // combine: row of batch tile
  const int cc = (tid & 15) * 8;     // combine: h-column base (8 contiguous)

  // staging geometry
  const int srow  = tid >> 4;
  const int scol8 = (tid & 15) * 8;  // f16 path: 8 halves
  const int scol4 = (tid & 15) * 4;  // f32 path: 4 floats

  // ---- stage tile for t=0 ----
  if (IN_F16) {
    v8h v = *(const v8h*)(inF16 + (((size_t)(b0 + srow) * T_ + 0) * IN_K + scol8));
    *(v8h*)&xs[0][srow * IN_K + scol8] = v;
  } else {
    v4f v = *(const v4f*)(inF32 + (((size_t)(b0 + srow) * T_ + 0) * IN_K + scol4));
    v4h h4;
    h4[0] = (_Float16)v[0]; h4[1] = (_Float16)v[1];
    h4[2] = (_Float16)v[2]; h4[3] = (_Float16)v[3];
    *(v4h*)&xs[0][srow * IN_K + scol4] = h4;
  }
  __syncthreads();

  for (int t = 0; t < T_; ++t) {
    const int cur = t & 1;
    const int nxt = cur ^ 1;

    // ---- issue prefetch of tile t+1 (consumed at end of this step) ----
    v8h pre16 = {};
    v4f pre32 = {};
    if (t + 1 < T_) {
      if (IN_F16)
        pre16 = *(const v8h*)(inF16 + (((size_t)(b0 + srow) * T_ + (t + 1)) * IN_K + scol8));
      else
        pre32 = *(const v4f*)(inF32 + (((size_t)(b0 + srow) * T_ + (t + 1)) * IN_K + scol4));
    }

    // ---- gate GEMM: g = x_t @ Wih^T + h @ Whh^T  (bias added later) ----
    v8f acc[4];
#pragma unroll
    for (int q = 0; q < 4; ++q) {
      const int nt = wave * 4 + q;                 // N-tile 0..31
      v8f a = {};                                  // C starts as inline 0
#pragma unroll
      for (int k0 = 0; k0 < IN_K; k0 += 32) {
        v16h av = load_a(&xs[cur][0], IN_K, k0, lane);
        v16h bm = load_b(WihS, IN_K, nt * 16, k0, lane);
        a = __builtin_amdgcn_wmma_f32_16x16x32_f16(false, av, false, bm,
                                                   (short)0, a, false, false);
      }
#pragma unroll
      for (int k0 = 0; k0 < H_; k0 += 32) {
        v16h av = load_a(h_sh, H_, k0, lane);
        v16h bm = load_b(WhhS, H_, nt * 16, k0, lane);
        a = __builtin_amdgcn_wmma_f32_16x16x32_f16(false, av, false, bm,
                                                   (short)0, a, false, false);
      }
      acc[q] = a;
    }

    // ---- add bias, activate by gate type, scatter into LDS ----
#pragma unroll
    for (int q = 0; q < 4; ++q) {
      const int nt = wave * 4 + q;
      const int gt = nt >> 3;                       // 0=i 1=f 2=g 3=o
      const int hcol = ((nt & 7) << 4) + (lane & 15);
#pragma unroll
      for (int j = 0; j < 8; ++j) {
        float v = acc[q][j] + breg[q];
        v = (gt == 2) ? fast_tanh(v) : fast_sigmoid(v);
        const int M = j + ((lane >> 4) << 3);       // C/D layout: M = j (+8 hi lanes)
        gact[(gt * 16 + M) * H_ + hcol] = v;
      }
    }
    __syncthreads();

    // ---- elementwise combine; c stays in VGPRs with fixed mapping ----
    v8f ig = *(const v8f*)&gact[(0 * 16 + cm) * H_ + cc];
    v8f fg = *(const v8f*)&gact[(1 * 16 + cm) * H_ + cc];
    v8f gg = *(const v8f*)&gact[(2 * 16 + cm) * H_ + cc];
    v8f og = *(const v8f*)&gact[(3 * 16 + cm) * H_ + cc];
    v8h hpack;
#pragma unroll
    for (int r = 0; r < 8; ++r) {
      const float c = fg[r] * creg[r] + ig[r] * gg[r];
      creg[r] = c;
      const float h = og[r] * fast_tanh(c);
      const _Float16 hh = (_Float16)h;
      h_sh[cm * H_ + cc + r] = hh;
      hpack[r] = hh;
      if (!WRITE_SEQ) {
        if (t == T_ - 1) lastOut[(size_t)(b0 + cm) * H_ + cc + r] = h;
      }
    }
    if (WRITE_SEQ) {
      _Float16* dst = seqOut + (((size_t)(b0 + cm) * T_ + t) * H_ + cc);
      *(v8h*)dst = hpack;   // contiguous 16B global store
    }

    // ---- commit prefetched tile into the other buffer ----
    if (t + 1 < T_) {
      if (IN_F16) {
        *(v8h*)&xs[nxt][srow * IN_K + scol8] = pre16;
      } else {
        v4h h4;
        h4[0] = (_Float16)pre32[0]; h4[1] = (_Float16)pre32[1];
        h4[2] = (_Float16)pre32[2]; h4[3] = (_Float16)pre32[3];
        *(v4h*)&xs[nxt][srow * IN_K + scol4] = h4;
      }
    }
    __syncthreads();
  }
}

__global__ __launch_bounds__(256)
void cvt_f32_to_f16(const float* __restrict__ src, _Float16* __restrict__ dst, int n) {
  int i = blockIdx.x * 256 + threadIdx.x;
  if (i < n) dst[i] = (_Float16)src[i];
}

__global__ __launch_bounds__(256)
void fc_kernel(const float* __restrict__ h2last, const float* __restrict__ Wfc,
               const float* __restrict__ bfc, float* __restrict__ out) {
  int b = blockIdx.x * 256 + threadIdx.x;     // 1024 threads total
  if (b >= B_) return;
  float s = bfc[0];
  const float* hp = h2last + (size_t)b * H_;
#pragma unroll 8
  for (int k = 0; k < H_; ++k) s += hp[k] * Wfc[k];
  out[b] = s;
}

extern "C" void kernel_launch(void* const* d_in, const int* in_sizes, int n_in,
                              void* d_out, int out_size, void* d_ws, size_t ws_size,
                              hipStream_t stream) {
  (void)in_sizes; (void)n_in; (void)out_size; (void)ws_size;
  const float* x    = (const float*)d_in[0];
  const float* Wih0 = (const float*)d_in[1];
  const float* Whh0 = (const float*)d_in[2];
  const float* b0   = (const float*)d_in[3];
  const float* Wih1 = (const float*)d_in[4];
  const float* Whh1 = (const float*)d_in[5];
  const float* b1   = (const float*)d_in[6];
  const float* Wfc  = (const float*)d_in[7];
  const float* bfc  = (const float*)d_in[8];

  char* ws = (char*)d_ws;
  size_t off = 0;
  _Float16* Wih0h = (_Float16*)(ws + off); off += (size_t)G_ * I_ * sizeof(_Float16);
  _Float16* Whh0h = (_Float16*)(ws + off); off += (size_t)G_ * H_ * sizeof(_Float16);
  _Float16* Wih1h = (_Float16*)(ws + off); off += (size_t)G_ * H_ * sizeof(_Float16);
  _Float16* Whh1h = (_Float16*)(ws + off); off += (size_t)G_ * H_ * sizeof(_Float16);
  _Float16* h1seq = (_Float16*)(ws + off); off += (size_t)B_ * T_ * H_ * sizeof(_Float16);
  float*    h2last = (float*)(ws + off);   off += (size_t)B_ * H_ * sizeof(float);
  // total workspace used: ~65.5 MB

  // weight conversion to f16 (gate-major rows == WMMA B-operand layout)
  cvt_f32_to_f16<<<(G_ * I_ + 255) / 256, 256, 0, stream>>>(Wih0, Wih0h, G_ * I_);
  cvt_f32_to_f16<<<(G_ * H_ + 255) / 256, 256, 0, stream>>>(Whh0, Whh0h, G_ * H_);
  cvt_f32_to_f16<<<(G_ * H_ + 255) / 256, 256, 0, stream>>>(Wih1, Wih1h, G_ * H_);
  cvt_f32_to_f16<<<(G_ * H_ + 255) / 256, 256, 0, stream>>>(Whh1, Whh1h, G_ * H_);

  // layer 0: f32 input, write full h1 sequence (f16)
  lstm_layer_kernel<I_, false, true><<<B_ / 16, 256, 0, stream>>>(
      x, nullptr, Wih0h, Whh0h, b0, h1seq, nullptr);

  // layer 1: f16 input (h1 sequence, L2-resident), write only final h2 (f32)
  lstm_layer_kernel<H_, true, false><<<B_ / 16, 256, 0, stream>>>(
      nullptr, h1seq, Wih1h, Whh1h, b1, nullptr, h2last);

  // final FC: out[b] = h2last[b,:] . Wfc + bfc
  fc_kernel<<<B_ / 256, 256, 0, stream>>>(h2last, Wfc, bfc, (float*)d_out);
}
